// PoetryModel_88691074662717
// MI455X (gfx1250) — compile-verified, matrix operand order checked by reading, not run on hardware
//
#include <hip/hip_runtime.h>
#include <math.h>

// ---------------------------------------------------------------------------
// Types for CDNA5 WMMA (wave32): bf16 A/B fragments, f32 accumulator
// ---------------------------------------------------------------------------
typedef __attribute__((ext_vector_type(16))) __bf16 bf16x16;
typedef __attribute__((ext_vector_type(8)))  __bf16 bf16x8;
typedef __attribute__((ext_vector_type(8)))  float  f32x8;

#define LSTM_B 64
#define LSTM_S 128
#define LSTM_V 8000
#define LSTM_E 256
#define LSTM_H 512
#define LSTM_G 2048  // 4*H

// ---------------------------------------------------------------------------
// Elementwise helpers
// ---------------------------------------------------------------------------
__global__ void cvt_f32_bf16_kernel(const float* __restrict__ src,
                                    __bf16* __restrict__ dst, int n) {
    int i = blockIdx.x * blockDim.x + threadIdx.x;
    int stride = gridDim.x * blockDim.x;
    for (; i < n; i += stride) dst[i] = (__bf16)src[i];
}

__global__ void bias_sum_kernel(const float* __restrict__ a,
                                const float* __restrict__ b,
                                float* __restrict__ out, int n) {
    int i = blockIdx.x * blockDim.x + threadIdx.x;
    if (i < n) out[i] = a[i] + b[i];
}

// Embedding gather fused with bf16 conversion: x_bf[(b*S+s)*E + e]
__global__ void gather_embed_kernel(const int* __restrict__ idx,
                                    const float* __restrict__ emb,
                                    __bf16* __restrict__ xbf, int M, int E) {
    int t = blockIdx.x * blockDim.x + threadIdx.x;
    int total = M * E;
    int stride = gridDim.x * blockDim.x;
    for (; t < total; t += stride) {
        int m = t >> 8;          // E = 256
        int e = t & (E - 1);
        xbf[t] = (__bf16)emb[(size_t)idx[m] * E + e];
    }
}

// ---------------------------------------------------------------------------
// Fragment loader per ISA 7.12.2 (16-bit A 16x32 layout; B mirrored, lanes=N).
//   lanes 0-15  (h=0): K = k0+0..7   and k0+16..23
//   lanes 16-31 (h=1): K = k0+8..15  and k0+24..31
// Two 16-byte loads per lane.
// ---------------------------------------------------------------------------
__device__ __forceinline__ bf16x16 load_frag(const __bf16* __restrict__ base,
                                             size_t row0, size_t ld, int k0) {
    unsigned lane = threadIdx.x & 31u;
    unsigned h = lane >> 4;
    unsigned r = lane & 15u;
    const __bf16* p = base + (row0 + r) * ld + (size_t)k0 + 8u * h;
    bf16x8 lo = *reinterpret_cast<const bf16x8*>(p);
    bf16x8 hi = *reinterpret_cast<const bf16x8*>(p + 16);
    bf16x16 out;
#pragma unroll
    for (int i = 0; i < 8; ++i) { out[i] = lo[i]; out[i + 8] = hi[i]; }
    return out;
}

// Same pattern, but from an LDS slab with fixed leading dim 32 (one K-slice).
// base points into a __shared__ array -> compiler emits ds_load_b128.
__device__ __forceinline__ bf16x16 load_frag_lds(const __bf16* base, int row0) {
    unsigned lane = threadIdx.x & 31u;
    unsigned h = lane >> 4;
    unsigned r = lane & 15u;
    const __bf16* p = base + (unsigned)(row0 + r) * 32u + 8u * h;
    bf16x8 lo = *reinterpret_cast<const bf16x8*>(p);
    bf16x8 hi = *reinterpret_cast<const bf16x8*>(p + 16);
    bf16x16 out;
#pragma unroll
    for (int i = 0; i < 8; ++i) { out[i] = lo[i]; out[i + 8] = hi[i]; }
    return out;
}

// ---------------------------------------------------------------------------
// CDNA5 async copy: one 16B lane-transfer global -> LDS, tracked by ASYNCcnt
// (ISA 15.18.3 GLOBAL_LOAD_ASYNC_TO_LDS_B128; generic LDS addr: low 32 bits).
// ---------------------------------------------------------------------------
__device__ __forceinline__ void async_g2l_b128(__bf16* lds_dst, const __bf16* gsrc) {
#if defined(__gfx1250__)
    unsigned ldsa = (unsigned)(uintptr_t)lds_dst;
    asm volatile("global_load_async_to_lds_b128 %0, %1, off"
                 :: "v"(ldsa), "v"(gsrc) : "memory");
#else
    *reinterpret_cast<bf16x8*>(lds_dst) = *reinterpret_cast<const bf16x8*>(gsrc);
#endif
}

__device__ __forceinline__ void wait_async(int pending) {
#if defined(__gfx1250__)
    if (pending)
        asm volatile("s_wait_asynccnt 0x1" ::: "memory");
    else
        asm volatile("s_wait_asynccnt 0x0" ::: "memory");
#endif
}

// ---------------------------------------------------------------------------
// Register-blocked inner product used by the non-cooperative GEMMs.
// ---------------------------------------------------------------------------
template <int MT, int NT>
__device__ __forceinline__ void wmma_block_mac(const __bf16* __restrict__ A, size_t lda,
                                               const __bf16* __restrict__ Bm, size_t ldb,
                                               size_t m0, size_t n0, int Ktiles,
                                               f32x8 (&acc)[MT][NT]) {
    for (int kt = 0; kt < Ktiles; ++kt) {
        const int k0 = kt * 32;
        bf16x16 a[MT], b[NT];
#pragma unroll
        for (int mt = 0; mt < MT; ++mt) a[mt] = load_frag(A, m0 + mt * 16, lda, k0);
#pragma unroll
        for (int nt = 0; nt < NT; ++nt) b[nt] = load_frag(Bm, n0 + nt * 16, ldb, k0);
#pragma unroll
        for (int mt = 0; mt < MT; ++mt)
#pragma unroll
            for (int nt = 0; nt < NT; ++nt)
                acc[mt][nt] = __builtin_amdgcn_wmma_f32_16x16x32_bf16(
                    false, a[mt], false, b[nt], (short)0, acc[mt][nt], false, false);
    }
}

// ---------------------------------------------------------------------------
// Blocked GEMM, plain row-major output:  D = A * B^T + bias[n]
// Macro-tile 32x64 per wave (MT=2, NT=4). Used for the input GEMM.
// ---------------------------------------------------------------------------
__global__ void wmma_gemm_blk_kernel(const __bf16* __restrict__ A, size_t lda,
                                     const __bf16* __restrict__ Bm, size_t ldb,
                                     const float* __restrict__ bias,
                                     float* __restrict__ D, size_t ldd,
                                     int Mblk, int Nblk, int Ktiles) {
    constexpr int MT = 2, NT = 4;
    int wid = threadIdx.x >> 5;
    int t = blockIdx.x * (blockDim.x >> 5) + wid;     // wave-uniform macro-tile id
    if (t >= Mblk * Nblk) return;                     // whole-wave exit: EXEC stays all-1s
    int tm = t / Nblk;
    int tn = t - tm * Nblk;
    size_t m0 = (size_t)tm * (MT * 16);
    size_t n0 = (size_t)tn * (NT * 16);

    f32x8 acc[MT][NT] = {};
    wmma_block_mac<MT, NT>(A, lda, Bm, ldb, m0, n0, Ktiles, acc);

    unsigned lane = threadIdx.x & 31u;
    unsigned h = lane >> 4;
    unsigned nn = lane & 15u;
#pragma unroll
    for (int nt = 0; nt < NT; ++nt) {
        size_t col = n0 + nt * 16 + nn;
        float bv = bias ? bias[col] : 0.f;
#pragma unroll
        for (int mt = 0; mt < MT; ++mt)
#pragma unroll
            for (int i = 0; i < 8; ++i) {
                size_t mrow = m0 + mt * 16 + (size_t)i + 8u * h;  // C/D layout: VGPR i -> M=i+8h
                D[mrow * ldd + col] = acc[mt][nt][i] + bv;
            }
    }
}

// ---------------------------------------------------------------------------
// Cooperative classifier GEMM (dominant 67 GFLOP):
//   block = 256 threads = 8 waves stacked in M; block tile 256(M) x 64(N).
//   W_cls K-slabs (64 rows x 32 K = 4 KB) are double-buffered in LDS via
//   GLOBAL_LOAD_ASYNC_TO_LDS_B128 (1 async b128 per thread per K-step,
//   issued one K-step ahead; s_wait_asynccnt + barrier before consumption).
//   Each wave: MT=2 x NT=4 register tile, A fragments from global/L2,
//   B fragments from LDS (ds_load_b128) shared by all 8 waves.
//   Epilogue scatters logits + b_cls directly into (B, V, S) layout.
// ---------------------------------------------------------------------------
__global__ void __launch_bounds__(256)
cls_gemm_coop_kernel(const __bf16* __restrict__ A,
                     const __bf16* __restrict__ Bm,
                     const float* __restrict__ bias,
                     float* __restrict__ out, int Nblk) {
    constexpr int MT = 2, NT = 4;
    constexpr int KT = LSTM_H / 32;          // 16 K-steps
    __shared__ __bf16 Bs[2][64 * 32];        // 2 x 4 KB double buffer

    int bm = blockIdx.x / Nblk;
    int bn = blockIdx.x - bm * Nblk;
    size_t m0blk = (size_t)bm * 256;
    size_t n0blk = (size_t)bn * 64;
    int wid = threadIdx.x >> 5;
    size_t m0 = m0blk + (size_t)wid * 32;

    // staging role of this thread: one b128 per K-step
    int srow = threadIdx.x >> 2;             // 0..63  (N row within slab)
    int sseg = threadIdx.x & 3;              // 0..3   (16B segment of 64B row)
    const __bf16* gB = Bm + (n0blk + srow) * (size_t)LSTM_H + sseg * 8;

    // prologue: stage K-step 0 into buffer 0
    async_g2l_b128(&Bs[0][srow * 32 + sseg * 8], gB);

    f32x8 acc[MT][NT] = {};
    for (int kt = 0; kt < KT; ++kt) {
        if (kt + 1 < KT) {   // stage next slab into the other buffer
            async_g2l_b128(&Bs[(kt + 1) & 1][srow * 32 + sseg * 8], gB + (kt + 1) * 32);
            wait_async(1);   // current slab's async (1 older op) complete
        } else {
            wait_async(0);
        }
        __syncthreads();     // all waves' slab data visible in LDS

        const __bf16* bs = Bs[kt & 1];
        bf16x16 a[MT], b[NT];
#pragma unroll
        for (int mt = 0; mt < MT; ++mt)
            a[mt] = load_frag(A, m0 + mt * 16, (size_t)LSTM_H, kt * 32);
#pragma unroll
        for (int nt = 0; nt < NT; ++nt)
            b[nt] = load_frag_lds(bs, nt * 16);
#pragma unroll
        for (int mt = 0; mt < MT; ++mt)
#pragma unroll
            for (int nt = 0; nt < NT; ++nt)
                acc[mt][nt] = __builtin_amdgcn_wmma_f32_16x16x32_bf16(
                    false, a[mt], false, b[nt], (short)0, acc[mt][nt], false, false);

        __syncthreads();     // all reads of this buffer done before re-staging
    }

    unsigned lane = threadIdx.x & 31u;
    unsigned h = lane >> 4;
    unsigned nn = lane & 15u;
#pragma unroll
    for (int nt = 0; nt < NT; ++nt) {
        size_t v = n0blk + nt * 16 + nn;               // vocab index
        float bv = bias[v];
#pragma unroll
        for (int mt = 0; mt < MT; ++mt)
#pragma unroll
            for (int i = 0; i < 8; ++i) {
                size_t mrow = m0 + mt * 16 + (size_t)i + 8u * h;  // mrow = b*S + s
                size_t bb = mrow >> 7;                             // S = 128
                size_t ss = mrow & 127u;
                out[((size_t)bb * LSTM_V + v) * LSTM_S + ss] = acc[mt][nt][i] + bv;
            }
    }
}

// ---------------------------------------------------------------------------
// Wave-per-tile GEMM kept for the tiny latency-bound recurrent step:
// D = A * B^T + Cadd      (maximum wave parallelism, 512 tiles)
// ---------------------------------------------------------------------------
__global__ void wmma_gemm_bf16_kernel(const __bf16* __restrict__ A, size_t lda,
                                      const __bf16* __restrict__ Bm, size_t ldb,
                                      const float* __restrict__ Cadd, size_t ldca,
                                      float* __restrict__ D, size_t ldd,
                                      int Mtiles, int Ntiles, int Ktiles) {
    int wid = threadIdx.x >> 5;
    int t = blockIdx.x * (blockDim.x >> 5) + wid;
    if (t >= Mtiles * Ntiles) return;
    int tm = t / Ntiles;
    int tn = t - tm * Ntiles;
    size_t m0 = (size_t)tm * 16;
    size_t n0 = (size_t)tn * 16;

    f32x8 acc = {0.f, 0.f, 0.f, 0.f, 0.f, 0.f, 0.f, 0.f};
    for (int kt = 0; kt < Ktiles; ++kt) {
        bf16x16 a = load_frag(A,  m0, lda, kt * 32);
        bf16x16 b = load_frag(Bm, n0, ldb, kt * 32);
        acc = __builtin_amdgcn_wmma_f32_16x16x32_bf16(
            false, a, false, b, (short)0, acc, false, false);
    }

    unsigned lane = threadIdx.x & 31u;
    unsigned h = lane >> 4;
    unsigned nn = lane & 15u;
#pragma unroll
    for (int i = 0; i < 8; ++i) {
        size_t mrow = m0 + (size_t)i + 8u * h;
        float v = acc[i] + Cadd[mrow * ldca + n0 + nn];
        D[mrow * ldd + n0 + nn] = v;
    }
}

// ---------------------------------------------------------------------------
// LSTM pointwise step: gates (B x 4H, gate order i,f,g,o) -> c,h update.
// h written as bf16 into hidden buffer at row stride S*H.
// ---------------------------------------------------------------------------
__global__ void lstm_act_kernel(const float* __restrict__ gates, size_t ldg,
                                float* __restrict__ cstate,
                                __bf16* __restrict__ h_out, size_t ldh) {
    int idx = blockIdx.x * blockDim.x + threadIdx.x;     // B*H threads
    if (idx >= LSTM_B * LSTM_H) return;
    int b = idx >> 9;             // H = 512
    int j = idx & (LSTM_H - 1);
    const float* gr = gates + (size_t)b * ldg;
    float gi = gr[j];
    float gf = gr[LSTM_H + j];
    float gg = gr[2 * LSTM_H + j];
    float go = gr[3 * LSTM_H + j];
    gi = 1.f / (1.f + __expf(-gi));
    gf = 1.f / (1.f + __expf(-gf));
    gg = tanhf(gg);
    go = 1.f / (1.f + __expf(-go));
    float c = gf * cstate[idx] + gi * gg;
    float hval = go * tanhf(c);
    cstate[idx] = c;
    h_out[(size_t)b * ldh + j] = (__bf16)hval;
}

// ---------------------------------------------------------------------------
// In-place log-softmax over V in (B, V, S) layout. Block = 128 threads (= S),
// one block per b; lane-contiguous s => coalesced 512B rows per v.
// Single online pass for (max, sumexp), one pass to write.
// ---------------------------------------------------------------------------
__global__ void logsoftmax_kernel(float* __restrict__ out) {
    int b = blockIdx.x;
    int s = threadIdx.x;
    float* p = out + (size_t)b * LSTM_V * LSTM_S + s;
    float m = -INFINITY, sum = 0.f;
    for (int v = 0; v < LSTM_V; ++v) {
        float x = p[(size_t)v * LSTM_S];
        float nm = fmaxf(m, x);
        sum = sum * __expf(m - nm) + __expf(x - nm);
        m = nm;
    }
    float lse = m + __logf(sum);
    for (int v = 0; v < LSTM_V; ++v) {
        p[(size_t)v * LSTM_S] -= lse;
    }
}

// ---------------------------------------------------------------------------
// Host launcher
// ---------------------------------------------------------------------------
extern "C" void kernel_launch(void* const* d_in, const int* in_sizes, int n_in,
                              void* d_out, int out_size, void* d_ws, size_t ws_size,
                              hipStream_t stream) {
    (void)in_sizes; (void)n_in; (void)out_size; (void)ws_size;
    const int B = LSTM_B, S = LSTM_S, V = LSTM_V, E = LSTM_E, H = LSTM_H, G = LSTM_G;

    const int*   inputs = (const int*)  d_in[0];
    const float* emb    = (const float*)d_in[1];
    const float* Wih    = (const float*)d_in[2];
    const float* Whh    = (const float*)d_in[3];
    const float* bih    = (const float*)d_in[4];
    const float* bhh    = (const float*)d_in[5];
    const float* Wcls   = (const float*)d_in[6];
    const float* bcls   = (const float*)d_in[7];
    float* out = (float*)d_out;

    // Workspace carve-out (256B aligned regions), total ~96 MB
    char* ws = (char*)d_ws;
    auto carve = [&](size_t bytes) {
        char* p = ws;
        ws += (bytes + 255) & ~(size_t)255;
        return p;
    };
    __bf16* Wih_bf  = (__bf16*)carve((size_t)G * E * 2);
    __bf16* Whh_bf  = (__bf16*)carve((size_t)G * H * 2);
    __bf16* Wcls_bf = (__bf16*)carve((size_t)V * H * 2);
    __bf16* x_bf    = (__bf16*)carve((size_t)B * S * E * 2);
    float*  xg      = (float*) carve((size_t)B * S * G * 4);
    __bf16* hid_bf  = (__bf16*)carve((size_t)B * S * H * 2);
    float*  gates   = (float*) carve((size_t)B * G * 4);
    float*  cst     = (float*) carve((size_t)B * H * 4);
    float*  bsum    = (float*) carve((size_t)G * 4);

    // 1) precision conversion + fused bias
    cvt_f32_bf16_kernel<<<2048, 256, 0, stream>>>(Wih,  Wih_bf,  G * E);
    cvt_f32_bf16_kernel<<<4096, 256, 0, stream>>>(Whh,  Whh_bf,  G * H);
    cvt_f32_bf16_kernel<<<8192, 256, 0, stream>>>(Wcls, Wcls_bf, V * H);
    bias_sum_kernel<<<(G + 255) / 256, 256, 0, stream>>>(bih, bhh, bsum, G);

    // 2) embedding gather (-> bf16)
    gather_embed_kernel<<<4096, 256, 0, stream>>>(inputs, emb, x_bf, B * S, E);

    // 3) input GEMM (blocked): xg = x_bf * Wih^T + (b_ih + b_hh)  [8192x2048, K=256]
    {
        int Mblk = (B * S) / 32, Nblk = G / 64, Ktiles = E / 32;   // 256 x 32 macro-tiles
        int blocks = (Mblk * Nblk + 7) / 8;                        // 8 waves / block
        wmma_gemm_blk_kernel<<<blocks, 256, 0, stream>>>(
            x_bf, (size_t)E, Wih_bf, (size_t)E,
            bsum, xg, (size_t)G, Mblk, Nblk, Ktiles);
    }

    // 4) sequential LSTM scan over S timesteps
    hipMemsetAsync(cst, 0, (size_t)B * H * 4, stream);
    for (int t = 0; t < S; ++t) {
        if (t == 0) {
            // gates = xg[:, 0, :]  (h0 == 0)
            lstm_act_kernel<<<(B * H + 255) / 256, 256, 0, stream>>>(
                xg, (size_t)S * G, cst, hid_bf, (size_t)S * H);
        } else {
            // gates = xg[:, t, :] + h_{t-1} * Whh^T     [64 x 2048, K=512]
            int Mtiles = B / 16, Ntiles = G / 16, Ktiles = H / 32;
            int blocks = (Mtiles * Ntiles + 7) / 8;   // 64 blocks, 512 waves
            wmma_gemm_bf16_kernel<<<blocks, 256, 0, stream>>>(
                hid_bf + (size_t)(t - 1) * H, (size_t)S * H,
                Whh_bf, (size_t)H,
                xg + (size_t)t * G, (size_t)S * G,
                gates, (size_t)G, Mtiles, Ntiles, Ktiles);
            lstm_act_kernel<<<(B * H + 255) / 256, 256, 0, stream>>>(
                gates, (size_t)G, cst,
                hid_bf + (size_t)t * H, (size_t)S * H);
        }
    }

    // 5) classifier GEMM (cooperative, async-LDS staged) into (B, V, S) + b_cls
    {
        int Mblk = (B * S) / 256;          // 32
        int Nblk = V / 64;                 // 125
        cls_gemm_coop_kernel<<<Mblk * Nblk, 256, 0, stream>>>(
            hid_bf, Wcls_bf, bcls, out, Nblk);
    }

    // 6) in-place log-softmax over V
    logsoftmax_kernel<<<B, S, 0, stream>>>(out);
}
